// Model_2121713844799
// MI455X (gfx1250) — compile-verified, matrix operand order checked by reading, not run on hardware
//
#include <hip/hip_runtime.h>
#include <hip/hip_bf16.h>
#include <math.h>

typedef __attribute__((ext_vector_type(16))) _Float16 v16h;
typedef __attribute__((ext_vector_type(8)))  float    v8f;

#define NWW 4096   // words per side
#define TCC 24     // chars per word
#define VCC 128    // char vocab
#define HH  32     // hidden
#define G4  128    // 4*H gates

// gfx1250 hardware V_TANH_F32 (confirmed in disasm): one TRANS op.
// sigmoid via the exact identity sig(x) = 0.5*tanh(x/2) + 0.5 -> 1 TRANS + 1 FMA.
__device__ __forceinline__ float fast_tanh(float x) {
#if __has_builtin(__builtin_amdgcn_tanhf)
    return __builtin_amdgcn_tanhf(x);
#elif __has_builtin(__builtin_amdgcn_tanh_f32)
    return __builtin_amdgcn_tanh_f32(x);
#else
    return tanhf(x);
#endif
}
__device__ __forceinline__ float sigf(float x) {
    return __builtin_fmaf(0.5f, fast_tanh(0.5f * x), 0.5f);
}

// ---------------------------------------------------------------------------
// Swizzled, bias-folded gather table:
//   P'[v*128 + (g&15)*8 + (g>>4)] = embed[v] . Wih[g]  +  b[g]
// so lane Lp's 8 C-input values for a word are 32B contiguous (2x b128 loads),
// and the per-step bias add disappears entirely.
// ---------------------------------------------------------------------------
__global__ void k_precompute_P(const float* __restrict__ embed,
                               const float* __restrict__ Wih,
                               const float* __restrict__ bias,
                               float* __restrict__ P) {
    int idx = blockIdx.x * blockDim.x + threadIdx.x;   // 0..16383
    int v = idx >> 7, g = idx & 127;
    float acc = bias[g];
#pragma unroll 8
    for (int k = 0; k < VCC; ++k) acc += embed[v * VCC + k] * Wih[g * VCC + k];
    P[v * G4 + (g & 15) * 8 + (g >> 4)] = acc;
}

// ---------------------------------------------------------------------------
// Mask-gated char LSTM over 24 steps. One wave = 16 words.
// Per step: gates[16w x 128g] = h[16w x 32] @ Whh^T  (+ (P'[char]) as WMMA C-in)
// K=32 -> exactly one v_wmma_f32_16x16x32_f16 per 16-gate tile (8 tiles).
// Mask-gating is an arithmetic blend (x += m*(new-x)) so clang cannot
// if-convert it back into exec-mask branches between WMMAs.
// ---------------------------------------------------------------------------
__global__ void __launch_bounds__(128)
k_char_lstm(const int* __restrict__ chars, const int* __restrict__ lens,
            const float* __restrict__ P, const float* __restrict__ Whh,
            float* __restrict__ seq_out) {
    __shared__ _Float16 hbuf[4][16][HH];
    const int wave = threadIdx.x >> 5, lane = threadIdx.x & 31;
    const int hlf = lane >> 4, Lp = lane & 15;
    const int w0 = blockIdx.x * 64 + wave * 16;

    for (int i = threadIdx.x; i < 4 * 16 * HH; i += blockDim.x)
        ((_Float16*)hbuf)[i] = (_Float16)0.f;

    // B fragments for Whh: tile n covers gates n*16 .. n*16+15 ; B[k][n] = Whh[g][k]
    v16h Bf[8];
#pragma unroll
    for (int n = 0; n < 8; ++n) {
        int g = n * 16 + Lp;
#pragma unroll
        for (int hx = 0; hx < 8; ++hx) {
            Bf[n][hx]     = (_Float16)Whh[g * HH + hlf * 8 + hx];
            Bf[n][8 + hx] = (_Float16)Whh[g * HH + 16 + hlf * 8 + hx];
        }
    }
    int len[8];
#pragma unroll
    for (int r = 0; r < 8; ++r) len[r] = lens[w0 + hlf * 8 + r];
    float cst[8][2], hst[8][2];
#pragma unroll
    for (int r = 0; r < 8; ++r)
        cst[r][0] = cst[r][1] = hst[r][0] = hst[r][1] = 0.f;

    for (int t = 0; t < TCC; ++t) {
        __syncthreads();
        // A fragment: M = Lp (word), K split per half-wave
        v16h a;
#pragma unroll
        for (int hx = 0; hx < 8; ++hx) {
            a[hx]     = hbuf[wave][Lp][hlf * 8 + hx];
            a[8 + hx] = hbuf[wave][Lp][16 + hlf * 8 + hx];
        }
        // C input = P'[char] (bias pre-folded), 2x b128 per word
        v8f acc[8];
#pragma unroll
        for (int r = 0; r < 8; ++r) {
            int ch = chars[(w0 + hlf * 8 + r) * TCC + t];
            const float4* Pr = (const float4*)(P + ch * G4 + Lp * 8);
            float4 p0 = Pr[0], p1 = Pr[1];
            acc[0][r] = p0.x; acc[1][r] = p0.y; acc[2][r] = p0.z; acc[3][r] = p0.w;
            acc[4][r] = p1.x; acc[5][r] = p1.y; acc[6][r] = p1.z; acc[7][r] = p1.w;
        }
#pragma unroll
        for (int n = 0; n < 8; ++n)
            acc[n] = __builtin_amdgcn_wmma_f32_16x16x32_f16(
                false, a, false, Bf[n], (short)0, acc[n], false, false);
        // LSTM cell; gate g = n*16+Lp : i -> n=0,1 ; f -> 2,3 ; g -> 4,5 ; o -> 6,7
#pragma unroll
        for (int r = 0; r < 8; ++r) {
            float m = (t < len[r]) ? 1.0f : 0.0f;   // v_cmp + v_cndmask, no branch
            int wl = hlf * 8 + r;
#pragma unroll
            for (int jj = 0; jj < 2; ++jj) {
                float ig = acc[0 + jj][r], fg = acc[2 + jj][r];
                float gg = acc[4 + jj][r], og = acc[6 + jj][r];
                float c2 = sigf(fg) * cst[r][jj] + sigf(ig) * fast_tanh(gg);
                float h2 = sigf(og) * fast_tanh(c2);
                cst[r][jj] = __builtin_fmaf(m, c2 - cst[r][jj], cst[r][jj]);
                hst[r][jj] = __builtin_fmaf(m, h2 - hst[r][jj], hst[r][jj]);
                hbuf[wave][wl][jj * 16 + Lp] = (_Float16)hst[r][jj];
            }
        }
    }
    __syncthreads();
    for (int i = lane; i < 16 * HH; i += 32) {
        int wl = i >> 5, j = i & 31;
        seq_out[(w0 + wl) * HH + j] = (float)hbuf[wave][wl][j];
    }
}

// ---------------------------------------------------------------------------
// preG[chain][t][g] = b[g] + seq[t] . Wih[g]   (all parallelizable work of the
// serial word LSTM, hoisted out of the 4096-step dependency chain)
// ---------------------------------------------------------------------------
__global__ void k_preG(const float* __restrict__ e_seq, const float* __restrict__ f_seq,
                       const float* W0, const float* W1, const float* W2, const float* W3,
                       const float* b0, const float* b1, const float* b2, const float* b3,
                       float* __restrict__ preG) {
    int idx = blockIdx.x * blockDim.x + threadIdx.x;   // 4 * 4096 * 128
    int chain = idx >> 19;
    int rem = idx & ((1 << 19) - 1);
    int t = rem >> 7, g = rem & 127;
    const float* seq = (chain < 2) ? e_seq : f_seq;
    const float* W = chain == 0 ? W0 : chain == 1 ? W1 : chain == 2 ? W2 : W3;
    const float* b = chain == 0 ? b0 : chain == 1 ? b1 : chain == 2 ? b2 : b3;
    float acc = b[g];
#pragma unroll
    for (int k = 0; k < HH; ++k) acc += seq[t * HH + k] * W[g * HH + k];
    preG[idx] = acc;
}

// ---------------------------------------------------------------------------
// Serial word-level LSTM chains (e_fwd, e_bwd, f_fwd, f_bwd) — latency-bound.
// 1 block of 128 threads per chain; thread g owns gate g (32 FMA/step);
// lanes 0..31 apply the cell (hardware tanh) and emit sigmoid(h) as f16.
// ---------------------------------------------------------------------------
__global__ void __launch_bounds__(128)
k_bilstm(const float* __restrict__ preG,
         const float* Wh0, const float* Wh1, const float* Wh2, const float* Wh3,
         _Float16* __restrict__ enc_e, _Float16* __restrict__ enc_f) {
    __shared__ float hs[HH];
    __shared__ float gl[G4];
    const int chain = blockIdx.x, g = threadIdx.x;
    const float* Wh = chain == 0 ? Wh0 : chain == 1 ? Wh1 : chain == 2 ? Wh2 : Wh3;
    _Float16* enc = (chain < 2) ? enc_e : enc_f;
    const int dir = chain & 1;
    const float* pg = preG + (size_t)chain * NWW * G4;
    float Wr[HH];
#pragma unroll
    for (int k = 0; k < HH; ++k) Wr[k] = Wh[g * HH + k];
    float creg = 0.f;
    if (g < HH) hs[g] = 0.f;
    __syncthreads();
    for (int s = 0; s < NWW; ++s) {
        int t = dir ? (NWW - 1 - s) : s;
        float acc = pg[t * G4 + g];
#pragma unroll
        for (int k = 0; k < HH; ++k) acc += Wr[k] * hs[k];
        gl[g] = acc;
        __syncthreads();
        if (g < HH) {
            float ig = gl[g], fg = gl[HH + g], gg = gl[2 * HH + g], og = gl[3 * HH + g];
            float c2 = sigf(fg) * creg + sigf(ig) * fast_tanh(gg);
            float h2 = sigf(og) * fast_tanh(c2);
            creg = c2;
            hs[g] = h2;
            enc[t * 64 + dir * HH + g] = (_Float16)sigf(h2);
        }
        __syncthreads();
    }
}

// ---------------------------------------------------------------------------
// S = f_enc @ e_enc^T  (4096 x 4096, K=64) — store-bandwidth bound.
// One wave = one 16x64 tile: 8 WMMAs, A-fragments reused 4x, and each output
// row's stores are 256B contiguous across the wave.
// ---------------------------------------------------------------------------
__global__ void __launch_bounds__(256)
k_score(const _Float16* __restrict__ fenc, const _Float16* __restrict__ eenc,
        float* __restrict__ out) {
    const int wave = threadIdx.x >> 5, lane = threadIdx.x & 31;
    const int hlf = lane >> 4, Lp = lane & 15;
    const int task = blockIdx.x * 8 + wave;       // 256 M-tiles x 64 N-groups
    const int i0 = (task >> 6) * 16, j0 = (task & 63) * 64;

    const _Float16* ar = fenc + (i0 + Lp) * 64;
    v16h a0, a1;
#pragma unroll
    for (int hx = 0; hx < 8; ++hx) {
        int k0 = hlf * 8 + hx, k1 = 16 + hlf * 8 + hx;
        a0[hx] = ar[k0];      a0[8 + hx] = ar[k1];
        a1[hx] = ar[32 + k0]; a1[8 + hx] = ar[32 + k1];
    }
    v8f acc[4];
#pragma unroll
    for (int nt = 0; nt < 4; ++nt) {
        const _Float16* br = eenc + (j0 + nt * 16 + Lp) * 64;
        v16h b0, b1;
#pragma unroll
        for (int hx = 0; hx < 8; ++hx) {
            int k0 = hlf * 8 + hx, k1 = 16 + hlf * 8 + hx;
            b0[hx] = br[k0];      b0[8 + hx] = br[k1];
            b1[hx] = br[32 + k0]; b1[8 + hx] = br[32 + k1];
        }
        v8f z = {};
        z = __builtin_amdgcn_wmma_f32_16x16x32_f16(false, a0, false, b0, (short)0, z, false, false);
        acc[nt] = __builtin_amdgcn_wmma_f32_16x16x32_f16(false, a1, false, b1, (short)0, z, false, false);
    }
#pragma unroll
    for (int r = 0; r < 8; ++r) {
        float* orow = out + (size_t)(i0 + hlf * 8 + r) * NWW + j0;
#pragma unroll
        for (int nt = 0; nt < 4; ++nt) orow[nt * 16 + Lp] = acc[nt][r];
    }
}

// ---------------------------------------------------------------------------
extern "C" void kernel_launch(void* const* d_in, const int* in_sizes, int n_in,
                              void* d_out, int out_size, void* d_ws, size_t ws_size,
                              hipStream_t stream) {
    const int* e_chars = (const int*)d_in[0];
    const int* e_lens  = (const int*)d_in[1];
    const int* f_chars = (const int*)d_in[2];
    const int* f_lens  = (const int*)d_in[3];
    // d_in[4] = diag (unused by reference)
    const float* embed_e = (const float*)d_in[5];
    const float* embed_f = (const float*)d_in[6];

    // workspace layout (floats)
    float* ws    = (float*)d_ws;
    float* P_e   = ws;                         // 128*128 (swizzled, bias-folded)
    float* P_f   = P_e + VCC * G4;             // 128*128
    float* e_seq = P_f + VCC * G4;             // 4096*32
    float* f_seq = e_seq + NWW * HH;           // 4096*32
    float* preG  = f_seq + NWW * HH;           // 4*4096*128
    _Float16* enc_e = (_Float16*)(preG + 4ull * NWW * G4);  // 4096*64 f16
    _Float16* enc_f = enc_e + NWW * 64;                     // 4096*64 f16

    // 1) collapse embedding-lookup @ Wih^T (+bias) into a swizzled gather table
    k_precompute_P<<<64, 256, 0, stream>>>(embed_e, (const float*)d_in[7],
                                           (const float*)d_in[9],  P_e);
    k_precompute_P<<<64, 256, 0, stream>>>(embed_f, (const float*)d_in[10],
                                           (const float*)d_in[12], P_f);

    // 2) WMMA char LSTMs (mask-gated) -> word features [4096, 32]
    k_char_lstm<<<64, 128, 0, stream>>>(e_chars, e_lens, P_e,
                                        (const float*)d_in[8],  e_seq);
    k_char_lstm<<<64, 128, 0, stream>>>(f_chars, f_lens, P_f,
                                        (const float*)d_in[11], f_seq);

    // 3) hoist input projections of the 4 serial chains (parallel)
    k_preG<<<8192, 256, 0, stream>>>(e_seq, f_seq,
        (const float*)d_in[13], (const float*)d_in[16], (const float*)d_in[19], (const float*)d_in[22],
        (const float*)d_in[15], (const float*)d_in[18], (const float*)d_in[21], (const float*)d_in[24],
        preG);

    // 4) serial recurrences (4 concurrent chains), emit sigmoid(enc) as f16
    k_bilstm<<<4, 128, 0, stream>>>(preG,
        (const float*)d_in[14], (const float*)d_in[17], (const float*)d_in[20], (const float*)d_in[23],
        enc_e, enc_f);

    // 5) 4096x4096 score matrix via WMMA (16x64 per wave)
    k_score<<<2048, 256, 0, stream>>>(enc_f, enc_e, (float*)d_out);
}
